// QORNN_Model_4028679323713
// MI455X (gfx1250) — compile-verified, matrix operand order checked by reading, not run on hardware
//
#include <hip/hip_runtime.h>
#include <hip/hip_fp16.h>
#include <math.h>

// ---------------------------------------------------------------------------
// QORNN eval on MI455X (gfx1250), wave32, WMMA f32_16x16x32_f16.
// B=256, T=128, I=128, H=1024, O=10.
// Recurrent part runs as ONE persistent kernel (32 blocks) with a device-wide
// monotonic spin barrier between timesteps instead of 128 dependent launches.
// ---------------------------------------------------------------------------

#define B_ 256
#define T_ 128
#define I_ 128
#define H_ 1024
#define O_ 10
#define EPS_ 1e-5f
#define RNN_BLOCKS 32

typedef __attribute__((ext_vector_type(16))) _Float16 v16h;
typedef __attribute__((ext_vector_type(8)))  _Float16 v8h;
typedef __attribute__((ext_vector_type(8)))  float    v8f;

// ---------------------------------------------------------------------------
// Fragment loader for 16-bit WMMA operands, row-major [rows x ld] f16 matrix.
// A operand: row = mTile + (lane&15), 16 f16 K-values per lane:
//   lane<16 : elems 0..7 -> K=k0+0..7,   elems 8..15 -> K=k0+16..23
//   lane>=16: elems 0..7 -> K=k0+8..15,  elems 8..15 -> K=k0+24..31
// B operand uses the same layout with "row" = the N index (weights stored
// row-major [N,K], i.e. Wr[j,k] directly — computes h @ Wr^T, no transpose).
// Two 16-byte loads -> global_load_b128.
// ---------------------------------------------------------------------------
__device__ __forceinline__ v16h load_frag16(const _Float16* __restrict__ base,
                                            int row, int ld, int k0, int lane) {
    int kb = k0 + ((lane & 16) ? 8 : 0);
    const _Float16* p = base + (size_t)row * ld + kb;
    v8h lo = *(const v8h*)(p);
    v8h hi = *(const v8h*)(p + 16);
    v16h r;
#pragma unroll
    for (int e = 0; e < 8; ++e) { r[e] = lo[e]; r[e + 8] = hi[e]; }
    return r;
}

__device__ __forceinline__ v8f wmma_f16(v16h a, v16h b, v8f c) {
    return __builtin_amdgcn_wmma_f32_16x16x32_f16(false, a, false, b,
                                                  (short)0, c, false, false);
}

// Device-wide barrier: monotonic counter, one arrival per block per phase.
__device__ __forceinline__ void grid_sync(unsigned* counter, unsigned target) {
    __syncthreads();
    if (threadIdx.x == 0) {
        __threadfence();  // make this block's h_out stores visible device-wide
        __hip_atomic_fetch_add(counter, 1u, __ATOMIC_ACQ_REL,
                               __HIP_MEMORY_SCOPE_AGENT);
        while (__hip_atomic_load(counter, __ATOMIC_ACQUIRE,
                                 __HIP_MEMORY_SCOPE_AGENT) < target) {
            __builtin_amdgcn_s_sleep(1);
        }
    }
    __syncthreads();
}

// ---------------------------------------------------------------------------
// Quantization: q = clamp(rint(x*s), -s, s-1)/s   (rint == round-half-even)
// ---------------------------------------------------------------------------
__global__ void quant_f16_kernel(const float* __restrict__ src,
                                 _Float16* __restrict__ dst,
                                 float scale, int n) {
    int i = blockIdx.x * blockDim.x + threadIdx.x;
    if (i >= n) return;
    float q = fminf(fmaxf(rintf(src[i] * scale), -scale), scale - 1.0f) * (1.0f / scale);
    dst[i] = (_Float16)q;
}

__global__ void quant_f32_kernel(const float* __restrict__ src,
                                 float* __restrict__ dst,
                                 float scale, int n) {
    int i = blockIdx.x * blockDim.x + threadIdx.x;
    if (i >= n) return;
    dst[i] = fminf(fmaxf(rintf(src[i] * scale), -scale), scale - 1.0f) * (1.0f / scale);
}

// ---------------------------------------------------------------------------
// Fold eval-norm + bias into per-feature affine: y = raw*s + o
//   s = g * rsqrt(rv+eps);  o = b + (bias - rm)*s   (bias=0 for output norm)
// ---------------------------------------------------------------------------
__global__ void prep_affine_kernel(const float* bi, const float* br,
                                   const float* ni_g, const float* ni_b,
                                   const float* ni_rm, const float* ni_rv,
                                   const float* nr_g, const float* nr_b,
                                   const float* nr_rm, const float* nr_rv,
                                   const float* no_g, const float* no_b,
                                   const float* no_rm, const float* no_rv,
                                   float* s_i, float* o_i,
                                   float* s_r, float* o_r,
                                   float* s_o, float* o_o) {
    int i = threadIdx.x;
    if (i < H_) {
        float si = ni_g[i] * rsqrtf(ni_rv[i] + EPS_);
        s_i[i] = si;
        o_i[i] = ni_b[i] + (bi[i] - ni_rm[i]) * si;
        float sr = nr_g[i] * rsqrtf(nr_rv[i] + EPS_);
        s_r[i] = sr;
        o_r[i] = nr_b[i] + (br[i] - nr_rm[i]) * sr;
    }
    if (i < O_) {
        float so = no_g[i] * rsqrtf(no_rv[i] + EPS_);
        s_o[i] = so;
        o_o[i] = no_b[i] - no_rm[i] * so;
    }
}

// ---------------------------------------------------------------------------
// Input GEMM: iact[t,b,h] = norm_i( xq[b,t,:] . Wi_q[h,:] + bi )  (folded affine)
// GEMM M = B*T = 32768 (rows = b*T+t), N = H, K = I = 128.
// One 16x16 tile per wave, 8 waves/block.
// ---------------------------------------------------------------------------
__global__ void input_gemm_kernel(const _Float16* __restrict__ xq,
                                  const _Float16* __restrict__ Wi_q,
                                  const float* __restrict__ s_i,
                                  const float* __restrict__ o_i,
                                  _Float16* __restrict__ iact) {
    const int lane = threadIdx.x & 31;
    const int wave = threadIdx.x >> 5;
    const int tile = blockIdx.x * 8 + wave;          // 131072 tiles total
    const int ntiles_n = H_ / 16;                    // 64
    const int mt = tile / ntiles_n;                  // 0..2047
    const int nt = tile % ntiles_n;                  // 0..63

    const int rowA = mt * 16 + (lane & 15);          // row in [B*T, I]
    const int rowB = nt * 16 + (lane & 15);          // row in [H, I]

    v8f acc = {};
#pragma unroll
    for (int k0 = 0; k0 < I_; k0 += 32) {
        v16h a = load_frag16(xq,   rowA, I_, k0, lane);
        v16h b = load_frag16(Wi_q, rowB, I_, k0, lane);
        acc = wmma_f16(a, b, acc);
    }

    const int n  = nt * 16 + (lane & 15);
    const float sc = s_i[n];
    const float of = o_i[n];
    const int mbase = mt * 16 + ((lane & 16) ? 8 : 0);
#pragma unroll
    for (int v = 0; v < 8; ++v) {
        int r = mbase + v;                 // flat row = b*T + t
        int b_ = r >> 7;                   // / T_  (T_ = 128)
        int t_ = r & (T_ - 1);             // % T_
        iact[((size_t)t_ * B_ + b_) * H_ + n] = (_Float16)(acc[v] * sc + of);
    }
}

// ---------------------------------------------------------------------------
// Persistent recurrent kernel: for t in 0..T-1:
//   h_out = tanh( iact_t + norm_r(h_in @ Wr_q^T + br) );  grid_sync; swap.
// GEMM per step: M = B = 256, N = H = 1024, K = H = 1024.
// Each wave owns one 16x64 strip (1 A-frag reused across 4 B-frags per K
// iteration, 32 K iterations). 8 waves/block, 32 blocks = 256 strips.
// ---------------------------------------------------------------------------
__global__ void rnn_persistent_kernel(const _Float16* __restrict__ Wr_q,
                                      const float* __restrict__ s_r,
                                      const float* __restrict__ o_r,
                                      const _Float16* __restrict__ iact,
                                      _Float16* __restrict__ hA,
                                      _Float16* __restrict__ hB,
                                      unsigned* __restrict__ barrier_ctr) {
    const int lane = threadIdx.x & 31;
    const int wave = threadIdx.x >> 5;
    const int strip = blockIdx.x * 8 + wave;         // 256 strips
    const int ngroups = H_ / 64;                     // 16
    const int mt = strip / ngroups;                  // 0..15
    const int ng = strip % ngroups;                  // 0..15

    const int rowA  = mt * 16 + (lane & 15);
    const int rowB0 = ng * 64 + (lane & 15);
    const int mbase = mt * 16 + ((lane & 16) ? 8 : 0);

    // Per-strip norm affine: constant across timesteps (one N per lane per j).
    float sc_j[4], of_j[4];
#pragma unroll
    for (int j = 0; j < 4; ++j) {
        const int n = ng * 64 + j * 16 + (lane & 15);
        sc_j[j] = s_r[n];
        of_j[j] = o_r[n];
    }

    const _Float16* h_in  = hA;   // zero-initialized h0
    _Float16*       h_out = hB;

    for (int t = 0; t < T_; ++t) {
        v8f acc0 = {}, acc1 = {}, acc2 = {}, acc3 = {};
        for (int k0 = 0; k0 < H_; k0 += 32) {
            v16h a  = load_frag16(h_in, rowA, H_, k0, lane);
            v16h b0 = load_frag16(Wr_q, rowB0,      H_, k0, lane);
            v16h b1 = load_frag16(Wr_q, rowB0 + 16, H_, k0, lane);
            v16h b2 = load_frag16(Wr_q, rowB0 + 32, H_, k0, lane);
            v16h b3 = load_frag16(Wr_q, rowB0 + 48, H_, k0, lane);
            acc0 = wmma_f16(a, b0, acc0);
            acc1 = wmma_f16(a, b1, acc1);
            acc2 = wmma_f16(a, b2, acc2);
            acc3 = wmma_f16(a, b3, acc3);
        }

        const _Float16* iact_t = iact + (size_t)t * B_ * H_;
        v8f accs[4] = {acc0, acc1, acc2, acc3};
#pragma unroll
        for (int j = 0; j < 4; ++j) {
            const int n = ng * 64 + j * 16 + (lane & 15);
#pragma unroll
            for (int v = 0; v < 8; ++v) {
                int m = mbase + v;
                float val = accs[j][v] * sc_j[j] + of_j[j]
                          + (float)iact_t[(size_t)m * H_ + n];
                h_out[(size_t)m * H_ + n] = (_Float16)tanhf(val);
            }
        }

        // All 32 blocks must finish step t before anyone reads h_out as h_in.
        grid_sync(barrier_ctr, (unsigned)(t + 1) * RNN_BLOCKS);

        const _Float16* tmp_in = h_out;
        h_out = (_Float16*)h_in;
        h_in  = tmp_in;
    }
}

// ---------------------------------------------------------------------------
// Output: out[b,o] = (h[b,:] . Wo_q[o,:]) * s_o[o] + o_o[o]
// ---------------------------------------------------------------------------
__global__ void out_gemm_kernel(const _Float16* __restrict__ h,
                                const float* __restrict__ Wo_q,
                                const float* __restrict__ s_o,
                                const float* __restrict__ o_o,
                                float* __restrict__ out) {
    int b = blockIdx.x * blockDim.x + threadIdx.x;
    if (b >= B_) return;
    const _Float16* hr = h + (size_t)b * H_;
    for (int o = 0; o < O_; ++o) {
        const float* wr = Wo_q + (size_t)o * H_;
        float s = 0.0f;
        for (int k = 0; k < H_; ++k) s += (float)hr[k] * wr[k];
        out[b * O_ + o] = s * s_o[o] + o_o[o];
    }
}

// ---------------------------------------------------------------------------
// Host-side launcher
// ---------------------------------------------------------------------------
static inline size_t align256(size_t x) { return (x + 255) & ~(size_t)255; }

extern "C" void kernel_launch(void* const* d_in, const int* in_sizes, int n_in,
                              void* d_out, int out_size, void* d_ws, size_t ws_size,
                              hipStream_t stream) {
    (void)in_sizes; (void)n_in; (void)out_size; (void)ws_size;

    const float* x     = (const float*)d_in[0];   // [B,T,I]
    const float* Wi    = (const float*)d_in[1];   // [H,I]
    const float* bi    = (const float*)d_in[2];
    const float* Wr    = (const float*)d_in[3];   // [H,H]
    const float* br    = (const float*)d_in[4];
    const float* Wo    = (const float*)d_in[5];   // [O,H]
    const float* ni_g  = (const float*)d_in[6];
    const float* ni_b  = (const float*)d_in[7];
    const float* ni_rm = (const float*)d_in[8];
    const float* ni_rv = (const float*)d_in[9];
    const float* nr_g  = (const float*)d_in[10];
    const float* nr_b  = (const float*)d_in[11];
    const float* nr_rm = (const float*)d_in[12];
    const float* nr_rv = (const float*)d_in[13];
    const float* no_g  = (const float*)d_in[14];
    const float* no_b  = (const float*)d_in[15];
    const float* no_rm = (const float*)d_in[16];
    const float* no_rv = (const float*)d_in[17];
    float* out = (float*)d_out;

    // Workspace carve-up
    char* ws = (char*)d_ws;
    size_t off = 0;
    _Float16* xq   = (_Float16*)(ws + off); off += align256((size_t)B_ * T_ * I_ * 2);
    _Float16* Wi_q = (_Float16*)(ws + off); off += align256((size_t)H_ * I_ * 2);
    _Float16* Wr_q = (_Float16*)(ws + off); off += align256((size_t)H_ * H_ * 2);
    float*    Wo_q = (float*)   (ws + off); off += align256((size_t)O_ * H_ * 4);
    float*    s_i  = (float*)   (ws + off); off += align256((size_t)H_ * 4);
    float*    o_i  = (float*)   (ws + off); off += align256((size_t)H_ * 4);
    float*    s_r  = (float*)   (ws + off); off += align256((size_t)H_ * 4);
    float*    o_r  = (float*)   (ws + off); off += align256((size_t)H_ * 4);
    float*    s_o  = (float*)   (ws + off); off += align256((size_t)O_ * 4);
    float*    o_o  = (float*)   (ws + off); off += align256((size_t)O_ * 4);
    _Float16* iact = (_Float16*)(ws + off); off += align256((size_t)T_ * B_ * H_ * 2);
    _Float16* hA   = (_Float16*)(ws + off); off += align256((size_t)B_ * H_ * 2);
    _Float16* hB   = (_Float16*)(ws + off); off += align256((size_t)B_ * H_ * 2);
    unsigned* bar  = (unsigned*)(ws + off); off += align256(256);

    // 1) Quantize inputs + weights (exact in f16: k/128 and k/8)
    {
        int n = B_ * T_ * I_;
        quant_f16_kernel<<<(n + 255) / 256, 256, 0, stream>>>(x, xq, 128.0f, n);
    }
    {
        int n = H_ * I_;
        quant_f16_kernel<<<(n + 255) / 256, 256, 0, stream>>>(Wi, Wi_q, 8.0f, n);
    }
    {
        int n = H_ * H_;
        quant_f16_kernel<<<(n + 255) / 256, 256, 0, stream>>>(Wr, Wr_q, 128.0f, n);
    }
    {
        int n = O_ * H_;
        quant_f32_kernel<<<(n + 255) / 256, 256, 0, stream>>>(Wo, Wo_q, 8.0f, n);
    }

    // 2) Fold norms/biases into per-feature affine
    prep_affine_kernel<<<1, 1024, 0, stream>>>(bi, br,
                                               ni_g, ni_b, ni_rm, ni_rv,
                                               nr_g, nr_b, nr_rm, nr_rv,
                                               no_g, no_b, no_rm, no_rv,
                                               s_i, o_i, s_r, o_r, s_o, o_o);

    // 3) Batched input GEMM over all timesteps (no sequential dependency)
    input_gemm_kernel<<<16384, 256, 0, stream>>>(xq, Wi_q, s_i, o_i, iact);

    // 4) h0 = 0; barrier counter = 0 (every call, for determinism)
    hipMemsetAsync(hA, 0, (size_t)B_ * H_ * 2, stream);
    hipMemsetAsync(bar, 0, sizeof(unsigned), stream);

    // 5) All 128 recurrent steps in ONE persistent kernel (32 co-resident
    //    blocks, device-wide barrier between steps) — removes 128 launch
    //    round-trips from the critical path.
    rnn_persistent_kernel<<<RNN_BLOCKS, 256, 0, stream>>>(Wr_q, s_r, o_r, iact,
                                                          hA, hB, bar);
    // After T=128 steps (even), final hidden state is back in hA.

    // 6) Output projection + norm
    out_gemm_kernel<<<1, 256, 0, stream>>>(hA, Wo_q, s_o, o_o, out);
}
